// GPT_11115375362051
// MI455X (gfx1250) — matrix operand order repaired and checked by reading, not compiled, not run on hardware
//
#include <hip/hip_runtime.h>
#include <hip/hip_bf16.h>
#include <math.h>

// ---------------------------------------------------------------------------
// GPT-2-small forward (B=2, T=1024, L=6, H=12, D=768, FF=3072, V=50257)
// GEMMs + flash-attention on v_wmma_f32_16x16x32_f16 (f16 in, f32 accum).
// GEMM B tiles staged with global_load_async_to_lds_b128 (+ s_wait_asynccnt).
// ---------------------------------------------------------------------------

#define Bsz   2
#define Tsz   1024
#define Lnum  6
#define Hnum  12
#define Dsz   768
#define HDsz  64
#define FFsz  3072
#define Vsz   50257
#define Vpad  50304          // 64*786, zero-padded columns for the head GEMM
#define EOSid 50256
#define BT    (Bsz*Tsz)      // 2048
#define BPAD  72             // halves per LDS B-tile row (144B, 16B aligned)
#define PPAD  40             // halves per LDS P-tile row  (80B, 16B aligned)

typedef _Float16 v16h __attribute__((ext_vector_type(16)));
typedef _Float16 v8h  __attribute__((ext_vector_type(8)));
typedef float    v8f  __attribute__((ext_vector_type(8)));

// ---- fragment packers ------------------------------------------------------
__device__ __forceinline__ v16h packAB(const _Float16* __restrict__ lo,
                                       const _Float16* __restrict__ hi) {
  v8h a0 = *(const v8h*)lo;
  v8h a1 = *(const v8h*)hi;
  v16h r;
#pragma unroll
  for (int i = 0; i < 8; ++i) { r[i] = a0[i]; r[8 + i] = a1[i]; }
  return r;
}

__device__ __forceinline__ float redmax16(float v) {
#pragma unroll
  for (int m = 1; m < 16; m <<= 1) v = fmaxf(v, __shfl_xor(v, m, 32));
  return v;
}
__device__ __forceinline__ float redsum16(float v) {
#pragma unroll
  for (int m = 1; m < 16; m <<= 1) v += __shfl_xor(v, m, 32);
  return v;
}

// ---- fp32 -> f16 bulk convert ---------------------------------------------
__global__ void k_cvt(const float* __restrict__ in, _Float16* __restrict__ out,
                      long n) {
  long i = (long)blockIdx.x * 256 + threadIdx.x;
  if (i < n) out[i] = (_Float16)in[i];
}

// ---- doc-id scan (document-boundary mask) ---------------------------------
__global__ void k_docid(const int* __restrict__ idx, int* __restrict__ doc) {
  int b = threadIdx.x;
  if (b < Bsz) {
    int acc = 0;
    for (int t = 0; t < Tsz; ++t) {
      int e = (idx[b*Tsz + t] == EOSid) ? 1 : 0;
      doc[b*Tsz + t] = acc;     // cumsum - is_eos == exclusive prefix
      acc += e;
    }
  }
}

// ---- embedding: x = tok[idx] + pos (fp32 residual stream) ------------------
__global__ void k_embed(const int* __restrict__ idx, const float* __restrict__ tok,
                        const float* __restrict__ pos, float* __restrict__ x) {
  int i = blockIdx.x * 256 + threadIdx.x;
  if (i >= BT * Dsz) return;
  int bt = i / Dsz, d = i - bt * Dsz, t = bt & (Tsz - 1);
  x[i] = tok[(size_t)idx[bt] * Dsz + d] + pos[t * Dsz + d];
}

// ---- LDS-tiled transpose of token_emb -> f16 embT[768][Vpad] ---------------
__global__ void k_embT(const float* __restrict__ emb, _Float16* __restrict__ embT) {
  __shared__ float tile[32][33];
  int n0 = blockIdx.x * 32;   // vocab dim
  int k0 = blockIdx.y * 32;   // feature dim
  for (int r = threadIdx.y; r < 32; r += 8) {
    int n = n0 + r, k = k0 + threadIdx.x;
    tile[r][threadIdx.x] = (n < Vsz) ? emb[(size_t)n * Dsz + k] : 0.f;
  }
  __syncthreads();
  for (int r = threadIdx.y; r < 32; r += 8) {
    int k = k0 + r, n = n0 + threadIdx.x;
    embT[(size_t)k * Vpad + n] = (_Float16)tile[threadIdx.x][r];
  }
}

// ---- K transpose: qkv16 K-part -> Kt[b,h,f(64),t(1024)] f16 ----------------
__global__ void k_kT(const _Float16* __restrict__ qkv, _Float16* __restrict__ kt) {
  __shared__ _Float16 tile[32][33];
  int bh = blockIdx.z;
  int b = bh / Hnum, h = bh - b * Hnum;
  int t0 = blockIdx.x * 32, f0 = blockIdx.y * 32;
  for (int r = threadIdx.y; r < 32; r += 8) {
    int t = t0 + r, f = f0 + threadIdx.x;
    tile[r][threadIdx.x] = qkv[(size_t)(b*Tsz + t) * (3*Dsz) + Dsz + h*HDsz + f];
  }
  __syncthreads();
  for (int r = threadIdx.y; r < 32; r += 8) {
    int f = f0 + r, t = t0 + threadIdx.x;
    kt[((size_t)(b*Hnum + h) * HDsz + f) * Tsz + t] = tile[threadIdx.x][r];
  }
}

// ---- layernorm (fp32 in -> f16 out, one 256-thread block per row) ----------
__global__ void k_ln(const float* __restrict__ in, _Float16* __restrict__ out,
                     const float* __restrict__ g, const float* __restrict__ b) {
  __shared__ float red[256];
  int row = blockIdx.x, tid = threadIdx.x;
  const float* p = in + (size_t)row * Dsz;
  float s = 0.f;
  for (int d = tid; d < Dsz; d += 256) s += p[d];
  red[tid] = s; __syncthreads();
  for (int k = 128; k > 0; k >>= 1) { if (tid < k) red[tid] += red[tid + k]; __syncthreads(); }
  float mu = red[0] * (1.f / Dsz); __syncthreads();
  float v = 0.f;
  for (int d = tid; d < Dsz; d += 256) { float t = p[d] - mu; v += t * t; }
  red[tid] = v; __syncthreads();
  for (int k = 128; k > 0; k >>= 1) { if (tid < k) red[tid] += red[tid + k]; __syncthreads(); }
  float rs = rsqrtf(red[0] * (1.f / Dsz) + 1e-5f);
  for (int d = tid; d < Dsz; d += 256)
    out[(size_t)row * Dsz + d] = (_Float16)((p[d] - mu) * rs * g[d] + b[d]);
}

// ---- WMMA GEMM: f16 A (global), f16 B (async-staged via LDS) ---------------
// grid = (N/64, M/64), block = 128 (4 waves). Wave w: rows [m0+16w, +16).
template<bool OUT16>
__global__ __launch_bounds__(128) void k_gemm(
    const _Float16* __restrict__ A, const _Float16* __restrict__ B,
    const float* __restrict__ bias, const float* __restrict__ resid,
    void* __restrict__ Cv, int K, int lda, int ldb, int ldc,
    int doGelu, int Nstore) {
  __shared__ _Float16 bt[32 * BPAD];
  const int tid  = threadIdx.x;
  const int lane = tid & 31;
  const int wave = tid >> 5;
  const int m0 = blockIdx.y * 64 + wave * 16;
  const int n0 = blockIdx.x * 64;
  const int g  = lane >> 4;
  const int ml = lane & 15;
  v8f acc[4] = {v8f{0}, v8f{0}, v8f{0}, v8f{0}};
  const _Float16* arow = A + (size_t)(m0 + ml) * lda + g * 8;
  const unsigned ldsbase = (unsigned)(size_t)(&bt[0]);
  const int r0 = tid >> 3;            // async chunk: row 0..15
  const int c0 = (tid & 7) * 8;       //              col offset (halves)
  for (int k0 = 0; k0 < K; k0 += 32) {
#pragma unroll
    for (int half = 0; half < 2; ++half) {
      int row = r0 + half * 16;
      const _Float16* gp = B + (size_t)(k0 + row) * ldb + n0 + c0;
      unsigned lp = ldsbase + (unsigned)(row * BPAD + c0) * 2;
      asm volatile("global_load_async_to_lds_b128 %0, %1, off"
                   :: "v"(lp), "v"((unsigned long long)(size_t)gp)
                   : "memory");
    }
    asm volatile("s_wait_asynccnt 0x0" ::: "memory");
    __syncthreads();
    v16h a = packAB(arow + k0, arow + k0 + 16);
    const _Float16* brow = &bt[(g * 8) * BPAD + (lane & 15)];
#pragma unroll
    for (int j = 0; j < 4; ++j) {
      v16h b;
#pragma unroll
      for (int i = 0; i < 8; ++i) { b[i] = brow[i * BPAD + 16 * j]; b[8 + i] = brow[(16 + i) * BPAD + 16 * j]; }
      acc[j] = __builtin_amdgcn_wmma_f32_16x16x32_f16(false, a, false, b,
                                                      (short)0, acc[j], false, false);
    }
    __syncthreads();   // before next k-step overwrites the tile
  }
  const int nl = lane & 15;
#pragma unroll
  for (int j = 0; j < 4; ++j) {
    int col = n0 + 16 * j + nl;
    if (col >= Nstore) continue;
    float bv = bias ? bias[col] : 0.f;
#pragma unroll
    for (int r = 0; r < 8; ++r) {
      int row = m0 + g * 8 + r;
      float v = acc[j][r] + bv;
      if (doGelu) v = 0.5f * v * (1.f + erff(v * 0.70710678118654752f));
      if constexpr (OUT16) {
        ((_Float16*)Cv)[(size_t)row * ldc + col] = (_Float16)v;
      } else {
        if (resid) v += resid[(size_t)row * ldc + col];
        ((float*)Cv)[(size_t)row * ldc + col] = v;
      }
    }
  }
}

// ---- WMMA flash-attention --------------------------------------------------
// grid = (T/64, H, B), block = 128 (4 independent waves).
// Wave w: 16 q-rows [q0, q0+16). Loops over 32-key tiles:
//   S(16x32) = Q(16x64) @ K^T  : 2 n-tiles x 2 chained WMMAs (Kt pre-transposed)
//   online softmax (row stats via 16-lane shuffles, lane holds 8 rows)
//   O(16x64) += P(16x32) @ V   : 4 WMMAs (P via wave-local LDS relayout)
__global__ __launch_bounds__(128) void k_fattn(
    const _Float16* __restrict__ qkv, const _Float16* __restrict__ kt,
    const int* __restrict__ doc, _Float16* __restrict__ y) {
  __shared__ _Float16 plds[4][16 * PPAD];
  const int tid = threadIdx.x, lane = tid & 31, wave = tid >> 5;
  const int h = blockIdx.y, b = blockIdx.z;
  const int q0 = blockIdx.x * 64 + wave * 16;
  const int g = lane >> 4, nl = lane & 15, ml = lane & 15;

  const _Float16* qbase = qkv + (size_t)(b*Tsz + q0 + ml) * (3*Dsz) + h*HDsz + g*8;
  v16h a0 = packAB(qbase, qbase + 16);       // features 0..31
  v16h a1 = packAB(qbase + 32, qbase + 48);  // features 32..63

  int qrow[8], docq[8];
#pragma unroll
  for (int r = 0; r < 8; ++r) {
    qrow[r] = q0 + g*8 + r;
    docq[r] = doc[b*Tsz + qrow[r]];
  }
  float mrow[8], lrow[8];
  v8f O[4] = {v8f{0}, v8f{0}, v8f{0}, v8f{0}};
#pragma unroll
  for (int r = 0; r < 8; ++r) { mrow[r] = -1e30f; lrow[r] = 0.f; }

  const int nkt = (q0 + 15) / 32 + 1;
  const _Float16* ktbase = kt + (size_t)(b*Hnum + h) * HDsz * Tsz;
  _Float16* pw = &plds[wave][0];

  for (int ktile = 0; ktile < nkt; ++ktile) {
    const int kbase = ktile * 32;
    v8f s[2];
#pragma unroll
    for (int nt = 0; nt < 2; ++nt) {
      const int t0 = kbase + nt * 16;
      const _Float16* kp0 = ktbase + (size_t)(g * 8) * Tsz + t0 + nl;         // f 0..31
      const _Float16* kp1 = ktbase + (size_t)(g * 8 + 32) * Tsz + t0 + nl;  // f 32..63
      v16h b0, b1;
#pragma unroll
      for (int i = 0; i < 8; ++i) {
        b0[i] = kp0[(size_t)i * Tsz]; b0[8 + i] = kp0[(size_t)(16 + i) * Tsz];
        b1[i] = kp1[(size_t)i * Tsz]; b1[8 + i] = kp1[(size_t)(16 + i) * Tsz];
      }
      v8f z = v8f{0};
      z = __builtin_amdgcn_wmma_f32_16x16x32_f16(false, a0, false, b0, (short)0, z, false, false);
      z = __builtin_amdgcn_wmma_f32_16x16x32_f16(false, a1, false, b1, (short)0, z, false, false);
      s[nt] = z;
    }
    // mask + scale (finite sentinel keeps online-softmax NaN-free)
#pragma unroll
    for (int nt = 0; nt < 2; ++nt) {
      int tk = kbase + nt * 16 + nl;
      int dock = doc[b*Tsz + tk];
#pragma unroll
      for (int r = 0; r < 8; ++r) {
        bool ok = (tk <= qrow[r]) && (dock == docq[r]);
        s[nt][r] = ok ? s[nt][r] * 0.125f : -1e30f;
      }
    }
    // online softmax per row (8 rows per lane)
    float alpha[8];
#pragma unroll
    for (int r = 0; r < 8; ++r) {
      float tm = redmax16(fmaxf(s[0][r], s[1][r]));
      float mnew = fmaxf(mrow[r], tm);
      alpha[r] = __expf(mrow[r] - mnew);
      mrow[r] = mnew;
      float p0 = __expf(s[0][r] - mnew);
      float p1 = __expf(s[1][r] - mnew);
      s[0][r] = p0; s[1][r] = p1;
      lrow[r] = lrow[r] * alpha[r] + redsum16(p0 + p1);
    }
    // D-layout -> A-layout exchange through wave-local LDS (in-order per wave)
#pragma unroll
    for (int nt = 0; nt < 2; ++nt)
#pragma unroll
      for (int r = 0; r < 8; ++r)
        pw[(g*8 + r) * PPAD + nt*16 + nl] = (_Float16)s[nt][r];
    const _Float16* pr = pw + ml * PPAD + g * 8;
    v16h pa = packAB(pr, pr + 16);
    // O update: rescale + P @ V
#pragma unroll
    for (int j = 0; j < 4; ++j) {
#pragma unroll
      for (int r = 0; r < 8; ++r) O[j][r] *= alpha[r];
      const _Float16* vp = qkv + (size_t)(b*Tsz + kbase + g*8) * (3*Dsz)
                           + 2*Dsz + h*HDsz + j*16 + nl;
      v16h bv;
#pragma unroll
      for (int i = 0; i < 8; ++i) { bv[i] = vp[(size_t)i * (3*Dsz)]; bv[8 + i] = vp[(size_t)(16 + i) * (3*Dsz)]; }
      O[j] = __builtin_amdgcn_wmma_f32_16x16x32_f16(false, pa, false, bv,
                                                    (short)0, O[j], false, false);
    }
  }
  // epilogue: y[row, h*64 + j*16 + nl] = O / l
#pragma unroll
  for (int r = 0; r < 8; ++r) {
    float inv = 1.f / lrow[r];
    size_t row = (size_t)(b*Tsz + qrow[r]);
#pragma unroll
    for (int j = 0; j < 4; ++j)
      y[row * Dsz + h*HDsz + j*16 + nl] = (_Float16)(O[j][r] * inv);
  }
}

// ---- loss ------------------------------------------------------------------
__global__ void k_zero(float* p) { if (threadIdx.x == 0 && blockIdx.x == 0) *p = 0.f; }

__global__ void k_loss(const float* __restrict__ logits, const int* __restrict__ tgt,
                       float* __restrict__ loss) {
  __shared__ float red[256];
  int row = blockIdx.x, tid = threadIdx.x;
  const float* p = logits + (size_t)row * Vsz;
  float mx = -3.0e38f;
  for (int j = tid; j < Vsz; j += 256) mx = fmaxf(mx, p[j]);
  red[tid] = mx; __syncthreads();
  for (int k = 128; k > 0; k >>= 1) { if (tid < k) red[tid] = fmaxf(red[tid], red[tid + k]); __syncthreads(); }
  mx = red[0]; __syncthreads();
  float s = 0.f;
  for (int j = tid; j < Vsz; j += 256) s += __expf(p[j] - mx);
  red[tid] = s; __syncthreads();
  for (int k = 128; k > 0; k >>= 1) { if (tid < k) red[tid] += red[tid + k]; __syncthreads(); }
  if (tid == 0) {
    float l = -(p[tgt[row]] - mx - logf(red[0]));
    atomicAdd(loss, l * (1.f / BT));
  }
}

// ---------------------------------------------------------------------------
extern "C" void kernel_launch(void* const* d_in, const int* in_sizes, int n_in,
                              void* d_out, int out_size, void* d_ws, size_t ws_size,
                              hipStream_t stream) {
  (void)in_sizes; (void)n_in; (void)out_size; (void)ws_size;
  const int*   idx   = (const int*)d_in[0];
  const int*   tgt   = (const int*)d_in[1];
  const float* tok   = (const float*)d_in[2];
  const float* pos   = (const float*)d_in[3];
  const float* ln1g  = (const float*)d_in[4];
  const float* ln1b  = (const float*)d_in[5];
  const float* ln2g  = (const float*)d_in[6];
  const float* ln2b  = (const float*)d_in[7];
  const float* Wqkv  = (const float*)d_in[8];
  const float* bqkv  = (const float*)d_in[9];
  const float* Wproj = (const float*)d_in[10];
  const float* bproj = (const float*)d_in[11];
  const float* Wfc1  = (const float*)d_in[12];
  const float* bfc1  = (const float*)d_in[13];
  const float* Wfc2  = (const float*)d_in[14];
  const float* bfc2  = (const float*)d_in[15];
  const float* lnfg  = (const float*)d_in[16];
  const float* lnfb  = (const float*)d_in[17];

  float* out = (float*)d_out;                      // [BT*V] logits, then [1] loss
  float* lossp = out + (size_t)BT * Vsz;

  char* ws = (char*)d_ws;
  size_t o = 0;
  float*    x    = (float*)(ws + o);    o += (size_t)BT * Dsz * 4;
  _Float16* h16  = (_Float16*)(ws + o); o += (size_t)BT * Dsz * 2;
  _Float16* qkv16= (_Float16*)(ws + o); o += (size_t)BT * 3 * Dsz * 2;
  _Float16* y16  = (_Float16*)(ws + o); o += (size_t)BT * Dsz * 2;
  _Float16* ff16 = (_Float16*)(ws + o); o += (size_t)BT * FFsz * 2;
  _Float16* kt16 = (_Float16*)(ws + o); o += (size_t)Bsz * Hnum * HDsz * Tsz * 2;
  int*      doc  = (int*)(ws + o);      o += (size_t)BT * 4;
  o = (o + 255) & ~(size_t)255;
  _Float16* embT = (_Float16*)(ws + o); o += (size_t)Dsz * Vpad * 2;
  _Float16* Wq16 = (_Float16*)(ws + o); o += (size_t)Lnum * Dsz * 3 * Dsz * 2;
  _Float16* Wp16 = (_Float16*)(ws + o); o += (size_t)Lnum * Dsz * Dsz * 2;
  _Float16* W116 = (_Float16*)(ws + o); o += (size_t)Lnum * Dsz * FFsz * 2;
  _Float16* W216 = (_Float16*)(ws + o); o += (size_t)Lnum * FFsz * Dsz * 2;

  // one-time (per launch) weight conversions fp32 -> f16
  {
    long nq = (long)Lnum * Dsz * 3 * Dsz;
    long np = (long)Lnum * Dsz * Dsz;
    long n1 = (long)Lnum * Dsz * FFsz;
    k_cvt<<<(unsigned)((nq + 255) / 256), 256, 0, stream>>>(Wqkv,  Wq16, nq);
    k_cvt<<<(unsigned)((np + 255) / 256), 256, 0, stream>>>(Wproj, Wp16, np);
    k_cvt<<<(unsigned)((n1 + 255) / 256), 256, 0, stream>>>(Wfc1,  W116, n1);
    k_cvt<<<(unsigned)((n1 + 255) / 256), 256, 0, stream>>>(Wfc2,  W216, n1);
  }
  k_docid<<<1, 32, 0, stream>>>(idx, doc);
  k_embed<<<(BT * Dsz + 255) / 256, 256, 0, stream>>>(idx, tok, pos, x);
  k_embT<<<dim3(Vpad / 32, Dsz / 32), dim3(32, 8), 0, stream>>>(tok, embT);

  for (int l = 0; l < Lnum; ++l) {
    k_ln<<<BT, 256, 0, stream>>>(x, h16, ln1g + l * Dsz, ln1b + l * Dsz);
    // qkv = h @ Wqkv + b          [2048 x 2304] -> f16
    k_gemm<true><<<dim3(3 * Dsz / 64, BT / 64), 128, 0, stream>>>(
        h16, Wq16 + (size_t)l * Dsz * 3 * Dsz, bqkv + (size_t)l * 3 * Dsz,
        nullptr, qkv16, Dsz, Dsz, 3 * Dsz, 3 * Dsz, 0, 3 * Dsz);
    k_kT<<<dim3(Tsz / 32, HDsz / 32, Bsz * Hnum), dim3(32, 8), 0, stream>>>(qkv16, kt16);
    k_fattn<<<dim3(Tsz / 64, Hnum, Bsz), 128, 0, stream>>>(qkv16, kt16, doc, y16);
    // x = x + y @ Wproj + b       [2048 x 768] -> fp32 residual stream
    k_gemm<false><<<dim3(Dsz / 64, BT / 64), 128, 0, stream>>>(
        y16, Wp16 + (size_t)l * Dsz * Dsz, bproj + (size_t)l * Dsz,
        x, x, Dsz, Dsz, Dsz, Dsz, 0, Dsz);
    k_ln<<<BT, 256, 0, stream>>>(x, h16, ln2g + l * Dsz, ln2b + l * Dsz);
    // ff = gelu(h @ Wfc1 + b)     [2048 x 3072] -> f16
    k_gemm<true><<<dim3(FFsz / 64, BT / 64), 128, 0, stream>>>(
        h16, W116 + (size_t)l * Dsz * FFsz, bfc1 + (size_t)l * FFsz,
        nullptr, ff16, Dsz, Dsz, FFsz, FFsz, 1, FFsz);
    // x = x + ff @ Wfc2 + b       [2048 x 768] -> fp32 residual stream
    k_gemm<false><<<dim3(Dsz / 64, BT / 64), 128, 0, stream>>>(
        ff16, W216 + (size_t)l * FFsz * Dsz, bfc2 + (size_t)l * Dsz,
        x, x, FFsz, FFsz, Dsz, Dsz, 0, Dsz);
  }

  k_ln<<<BT, 256, 0, stream>>>(x, h16, lnfg, lnfb);
  // logits = h @ embT             [2048 x 50257], zero-padded f16 B
  k_gemm<false><<<dim3(Vpad / 64, BT / 64), 128, 0, stream>>>(
      h16, embT, nullptr, nullptr, out, Dsz, Dsz, Vpad, Vsz, 0, Vsz);

  k_zero<<<1, 1, 0, stream>>>(lossp);
  k_loss<<<BT, 256, 0, stream>>>(out, tgt, lossp);
}
